// MoE_74852690035269
// MI455X (gfx1250) — compile-verified, matrix operand order checked by reading, not run on hardware
//
#include <hip/hip_runtime.h>
#include <math.h>

#define IN_DIM 512
#define HID    1024
#define NEXP   16
#define NTOK   8192
#define MT     64                       // token rows per expert tile
#define PADS   (2*NTOK + NEXP*MT)       // padded slot array size

typedef __attribute__((ext_vector_type(16))) _Float16 v16h;
typedef __attribute__((ext_vector_type(8)))  float    v8f;

union F16Frag { v16h h; uint4 u[2]; };

// Load a 16x32 f16 fragment (A layout per CDNA5 ISA 7.12.2) from a row-major
// matrix with leading dimension `ld` (elements). Works for LDS or global.
// lane 0-15: row = base+lane, K = [0..7],[16..23]; lane 16-31: K = [8..15],[24..31]
__device__ __forceinline__ v16h load_frag(const _Float16* base, int ld, int rbase,
                                          int k0, int lane) {
    int r  = rbase + (lane & 15);
    int kb = k0 + ((lane >> 4) << 3);
    const _Float16* p = base + (size_t)r * ld + kb;
    F16Frag f;
    f.u[0] = *(const uint4*)(p);
    f.u[1] = *(const uint4*)(p + 16);
    return f.h;
}

__device__ __forceinline__ v8f wmma_f16(v16h a, v16h b, v8f c) {
    return __builtin_amdgcn_wmma_f32_16x16x32_f16(false, a, false, b,
                                                  (short)0, c, false, false);
}

__device__ __forceinline__ float gelu_exact(float v) {
    return 0.5f * v * (1.0f + erff(v * 0.70710678118654752f));
}

// ---------------- weight transpose + f32->f16 convert ----------------
// in:  [E][R][C] f32 (row-major, K-major)   out: [E][C][R] f16 (N-major)
__global__ __launch_bounds__(256) void transpose_cvt_kernel(
    const float* __restrict__ in, _Float16* __restrict__ outp, int R, int C) {
    __shared__ float tile[32][33];
    int e = blockIdx.z;
    const float* ine  = in   + (size_t)e * R * C;
    _Float16*    oute = outp + (size_t)e * R * C;
    int c0 = blockIdx.x * 32, r0 = blockIdx.y * 32;
    int tx = threadIdx.x & 31, ty = threadIdx.x >> 5;   // ty: 0..7
    #pragma unroll
    for (int i = 0; i < 4; ++i)
        tile[ty + i*8][tx] = ine[(size_t)(r0 + ty + i*8) * C + c0 + tx];
    __syncthreads();
    #pragma unroll
    for (int i = 0; i < 4; ++i)
        oute[(size_t)(c0 + ty + i*8) * R + r0 + tx] = (_Float16)tile[tx][ty + i*8];
}

// ---------------- zero counters / token lists ----------------
__global__ void zero_kernel(int* counts, int* cursor, int* tokenIdx, float* tokenW) {
    int i = blockIdx.x * 256 + threadIdx.x;
    if (i < NEXP) { counts[i] = 0; cursor[i] = 0; }
    if (i < PADS) { tokenIdx[i] = 0; tokenW[i] = 0.0f; }
}

// ---------------- out = x (residual) ----------------
__global__ void copy_residual_kernel(const float* __restrict__ x,
                                     float* __restrict__ out, int n4) {
    int i = blockIdx.x * 256 + threadIdx.x;
    if (i < n4) ((float4*)out)[i] = ((const float4*)x)[i];
}

// ---------------- router: one wave per token ----------------
__global__ __launch_bounds__(256) void router_kernel(
    const float* __restrict__ x, const float* __restrict__ Wr,
    const float* __restrict__ br, int* __restrict__ counts,
    int* __restrict__ te, float* __restrict__ tw) {
    int lane = threadIdx.x & 31, wid = threadIdx.x >> 5;
    int t = blockIdx.x * 8 + wid;
    if (t >= NTOK) return;
    float acc[NEXP];
    #pragma unroll
    for (int e = 0; e < NEXP; ++e) acc[e] = 0.0f;
    const float* xr = x + (size_t)t * IN_DIM;
    for (int d = lane; d < IN_DIM; d += 32) {
        float xv = xr[d];
        const float* wr = Wr + d * NEXP;
        #pragma unroll
        for (int e = 0; e < NEXP; ++e) acc[e] += xv * wr[e];
    }
    #pragma unroll
    for (int off = 16; off >= 1; off >>= 1) {
        #pragma unroll
        for (int e = 0; e < NEXP; ++e) acc[e] += __shfl_down(acc[e], off, 32);
    }
    if (lane == 0) {
        #pragma unroll
        for (int e = 0; e < NEXP; ++e) acc[e] += br[e];
        float l0 = -3.4e38f; int e0 = 0;
        #pragma unroll
        for (int e = 0; e < NEXP; ++e) if (acc[e] > l0) { l0 = acc[e]; e0 = e; }
        float l1 = -3.4e38f; int e1 = 0;
        #pragma unroll
        for (int e = 0; e < NEXP; ++e)
            if (e != e0 && acc[e] > l1) { l1 = acc[e]; e1 = e; }
        // top-2 softmax weights renormalized: Z cancels
        float r  = __expf(l1 - l0);
        float w0 = 1.0f / (1.0f + r);
        float w1 = r    / (1.0f + r);
        te[t*2] = e0; te[t*2+1] = e1;
        tw[t*2] = w0; tw[t*2+1] = w1;
        atomicAdd(&counts[e0], 1);
        atomicAdd(&counts[e1], 1);
    }
}

// ---------------- prefix scan over experts (tiny) ----------------
__global__ void scan_kernel(const int* __restrict__ counts, int* __restrict__ slotOff,
                            int* __restrict__ tileOff, int* __restrict__ totalTiles) {
    if (threadIdx.x == 0 && blockIdx.x == 0) {
        int so = 0, to = 0;
        for (int e = 0; e < NEXP; ++e) {
            slotOff[e] = so; tileOff[e] = to;
            int tiles = (counts[e] + MT - 1) / MT;
            so += tiles * MT; to += tiles;
        }
        tileOff[NEXP] = to;
        *totalTiles = to;
    }
}

// ---------------- scatter tokens into per-expert lists ----------------
__global__ void scatter_kernel(const int* __restrict__ te, const float* __restrict__ tw,
                               const int* __restrict__ slotOff, int* __restrict__ cursor,
                               int* __restrict__ tokenIdx, float* __restrict__ tokenW) {
    int t = blockIdx.x * 256 + threadIdx.x;
    if (t >= NTOK) return;
    #pragma unroll
    for (int k = 0; k < 2; ++k) {
        int e = te[t*2 + k];
        int pos = atomicAdd(&cursor[e], 1);
        int s = slotOff[e] + pos;
        tokenIdx[s] = t;
        tokenW[s]   = tw[t*2 + k];
    }
}

// ---------------- fused expert MLP (WMMA) ----------------
__global__ __launch_bounds__(256, 1) void moe_expert_kernel(
    const float* __restrict__ x,
    const _Float16* __restrict__ W1t,   // [E][HID][IN_DIM] f16, N-major
    const float* __restrict__ b1,       // [E][HID]
    const _Float16* __restrict__ W2t,   // [E][IN_DIM][HID] f16, N-major
    const float* __restrict__ b2,       // [E][IN_DIM]
    const int* __restrict__ tokenIdx, const float* __restrict__ tokenW,
    const int* __restrict__ slotOff, const int* __restrict__ tileOff,
    const int* __restrict__ totalTiles,
    float* __restrict__ out) {

    __shared__ _Float16 ldsX[MT][IN_DIM];   // 64 KB
    __shared__ _Float16 ldsH[MT][HID];      // 128 KB
    __shared__ int   tokS[MT];
    __shared__ float wS[MT];

    const int tid  = threadIdx.x;
    const int lane = tid & 31;
    const int wid  = tid >> 5;
    const int nt   = *totalTiles;
    const int coll = lane & 15;
    const int rsel = (lane >> 4) << 3;

    for (int tile = blockIdx.x; tile < nt; tile += gridDim.x) {
        __syncthreads();   // protect LDS reuse across tiles

        // map tile -> expert
        int e = 0;
        #pragma unroll
        for (int q = 0; q < NEXP; ++q) if (tile >= tileOff[q + 1]) e = q + 1;
        int mtile = tile - tileOff[e];
        int slot0 = slotOff[e] + mtile * MT;

        if (tid < MT) { tokS[tid] = tokenIdx[slot0 + tid]; wS[tid] = tokenW[slot0 + tid]; }
        __syncthreads();

        // gather X tile (f32 -> f16) into LDS
        for (int i = tid; i < MT * IN_DIM; i += 256) {
            int r = i >> 9, c = i & (IN_DIM - 1);
            ldsX[r][c] = (_Float16)x[(size_t)tokS[r] * IN_DIM + c];
        }
        __syncthreads();

        const _Float16* w1e = W1t + (size_t)e * HID * IN_DIM;
        const float*    b1e = b1 + e * HID;

        // ---- GEMM1: H = gelu(X[64x512] @ W1 + b1) -> ldsH[64][1024] ----
        #pragma unroll 1
        for (int np = 0; np < 2; ++np) {
            int nbase = wid * 64 + np * 512;
            v8f c[4][4];
            v8f z = {0.f,0.f,0.f,0.f,0.f,0.f,0.f,0.f};
            #pragma unroll
            for (int mi = 0; mi < 4; ++mi)
                #pragma unroll
                for (int ni = 0; ni < 4; ++ni) c[mi][ni] = z;

            #pragma unroll 1
            for (int k0 = 0; k0 < IN_DIM; k0 += 32) {
                v16h a[4], b[4];
                #pragma unroll
                for (int mi = 0; mi < 4; ++mi)
                    a[mi] = load_frag(&ldsX[0][0], IN_DIM, mi * 16, k0, lane);
                #pragma unroll
                for (int ni = 0; ni < 4; ++ni)
                    b[ni] = load_frag(w1e, IN_DIM, nbase + ni * 16, k0, lane);
                #pragma unroll
                for (int mi = 0; mi < 4; ++mi)
                    #pragma unroll
                    for (int ni = 0; ni < 4; ++ni)
                        c[mi][ni] = wmma_f16(a[mi], b[ni], c[mi][ni]);
            }
            // epilogue: +b1, exact GELU, store f16 to ldsH
            #pragma unroll
            for (int ni = 0; ni < 4; ++ni) {
                int col = nbase + ni * 16 + coll;
                float bias = b1e[col];
                #pragma unroll
                for (int mi = 0; mi < 4; ++mi)
                    #pragma unroll
                    for (int j = 0; j < 8; ++j) {
                        int row = mi * 16 + rsel + j;
                        float v = c[mi][ni][j] + bias;
                        ldsH[row][col] = (_Float16)gelu_exact(v);
                    }
            }
        }
        __syncthreads();

        const _Float16* w2e = W2t + (size_t)e * IN_DIM * HID;
        const float*    b2e = b2 + e * IN_DIM;

        // ---- GEMM2: O = H[64x1024] @ W2 + b2; weighted atomic scatter ----
        {
            int nbase = wid * 64;
            v8f c[4][4];
            v8f z = {0.f,0.f,0.f,0.f,0.f,0.f,0.f,0.f};
            #pragma unroll
            for (int mi = 0; mi < 4; ++mi)
                #pragma unroll
                for (int ni = 0; ni < 4; ++ni) c[mi][ni] = z;

            #pragma unroll 1
            for (int k0 = 0; k0 < HID; k0 += 32) {
                v16h a[4], b[4];
                #pragma unroll
                for (int mi = 0; mi < 4; ++mi)
                    a[mi] = load_frag(&ldsH[0][0], HID, mi * 16, k0, lane);
                #pragma unroll
                for (int ni = 0; ni < 4; ++ni)
                    b[ni] = load_frag(w2e, HID, nbase + ni * 16, k0, lane);
                #pragma unroll
                for (int mi = 0; mi < 4; ++mi)
                    #pragma unroll
                    for (int ni = 0; ni < 4; ++ni)
                        c[mi][ni] = wmma_f16(a[mi], b[ni], c[mi][ni]);
            }
            #pragma unroll
            for (int ni = 0; ni < 4; ++ni) {
                int col = nbase + ni * 16 + coll;
                float bias = b2e[col];
                #pragma unroll
                for (int mi = 0; mi < 4; ++mi)
                    #pragma unroll
                    for (int j = 0; j < 8; ++j) {
                        int row = mi * 16 + rsel + j;
                        float w = wS[row];
                        if (w != 0.0f) {
                            float val = w * (c[mi][ni][j] + bias);
                            atomicAdd(&out[(size_t)tokS[row] * IN_DIM + col], val);
                        }
                    }
            }
        }
    }
}

extern "C" void kernel_launch(void* const* d_in, const int* in_sizes, int n_in,
                              void* d_out, int out_size, void* d_ws, size_t ws_size,
                              hipStream_t stream) {
    (void)in_sizes; (void)n_in; (void)out_size; (void)ws_size;
    const float* x  = (const float*)d_in[0];
    const float* Wr = (const float*)d_in[1];
    const float* br = (const float*)d_in[2];
    const float* W1 = (const float*)d_in[3];
    const float* b1 = (const float*)d_in[4];
    const float* W2 = (const float*)d_in[5];
    const float* b2 = (const float*)d_in[6];
    float* out = (float*)d_out;

    // workspace layout
    char* ws = (char*)d_ws;
    _Float16* W1t = (_Float16*)ws;                       // 16 MB
    _Float16* W2t = (_Float16*)(ws + (size_t)16777216);  // 16 MB
    char* p = ws + (size_t)2 * 16777216;
    int*   counts     = (int*)p;   p += 64;
    int*   cursor     = (int*)p;   p += 64;
    int*   slotOff    = (int*)p;   p += 64;
    int*   tileOff    = (int*)p;   p += 128;   // 17 ints
    int*   totalTiles = (int*)p;   p += 64;
    int*   te         = (int*)p;   p += (size_t)NTOK * 2 * 4;
    float* tw         = (float*)p; p += (size_t)NTOK * 2 * 4;
    int*   tokenIdx   = (int*)p;   p += (size_t)PADS * 4;
    float* tokenW     = (float*)p;

    // 1) weights -> f16, N-major (L2-resident thereafter)
    transpose_cvt_kernel<<<dim3(HID/32, IN_DIM/32, NEXP), 256, 0, stream>>>(W1, W1t, IN_DIM, HID);
    transpose_cvt_kernel<<<dim3(IN_DIM/32, HID/32, NEXP), 256, 0, stream>>>(W2, W2t, HID, IN_DIM);
    // 2) zero counters & lists; out = x (residual)
    zero_kernel<<<(PADS + 255) / 256, 256, 0, stream>>>(counts, cursor, tokenIdx, tokenW);
    copy_residual_kernel<<<(NTOK * IN_DIM / 4 + 255) / 256, 256, 0, stream>>>(x, out, NTOK * IN_DIM / 4);
    // 3) router
    router_kernel<<<NTOK / 8, 256, 0, stream>>>(x, Wr, br, counts, te, tw);
    // 4) scan + scatter
    scan_kernel<<<1, 32, 0, stream>>>(counts, slotOff, tileOff, totalTiles);
    scatter_kernel<<<NTOK / 256, 256, 0, stream>>>(te, tw, slotOff, cursor, tokenIdx, tokenW);
    // 5) fused expert GEMMs (persistent; max tiles = 2*8192/64 + 16 = 272)
    moe_expert_kernel<<<288, 256, 0, stream>>>(x, W1t, b1, W2t, b2,
                                               tokenIdx, tokenW, slotOff, tileOff,
                                               totalTiles, out);
}
